// LeViTAttentionSubsampleBlock_48833778155729
// MI455X (gfx1250) — matrix operand order repaired and checked by reading, not compiled, not run on hardware
//
#include <hip/hip_runtime.h>

// ---------------------------------------------------------------------------
// LeViT attention subsample block for MI455X (gfx1250), bf16 WMMA pipeline.
// ---------------------------------------------------------------------------

#define B_DIM      64
#define RES_IN     28
#define RES_OUT    14
#define SEQ        784          // 28*28
#define SEQ_OUT    196          // 14*14
#define C_IN       512
#define KEY_DIM    32
#define NUM_HEADS  16
#define D_V        128
#define OUT_KV     2560         // (32+128)*16
#define OUT_Q      512          // 32*16
#define OUT_PROJ   2048         // 128*16
#define C_OUT      768
#define SCALE_ATT  0.17677669529663688f   // 32^-0.5
#define BN_EPS     1e-5f

#define M_KV   (B_DIM * SEQ)      // 50176
#define M_Q    (B_DIM * SEQ_OUT)  // 12544
#define SEQ_P  800                // padded key length (25 * 32)
#define QT_TILES 13               // ceil(196/16)

typedef __bf16 bf16_t;
typedef __attribute__((ext_vector_type(16))) __bf16     v16bf;
typedef __attribute__((ext_vector_type(8)))  float      v8f;
typedef __attribute__((ext_vector_type(4)))  unsigned   u32x4;

union Frag {
    v16bf  v;
    u32x4  u[2];
    bf16_t e[16];
};

static __device__ inline v8f wmma_bf16(const Frag& a, const Frag& b, v8f c) {
    return __builtin_amdgcn_wmma_f32_16x16x32_bf16(false, a.v, false, b.v,
                                                   (short)0, c, false, false);
}

// ---------------------------------------------------------------------------
// Prep kernels
// ---------------------------------------------------------------------------

__global__ void bn_fuse_kernel(const float* __restrict__ g, const float* __restrict__ bb,
                               const float* __restrict__ m, const float* __restrict__ v,
                               float* __restrict__ scale, float* __restrict__ bias, int n) {
    int i = blockIdx.x * blockDim.x + threadIdx.x;
    if (i < n) {
        float s = g[i] * rsqrtf(v[i] + BN_EPS);
        scale[i] = s;
        bias[i]  = bb[i] - m[i] * s;
    }
}

__global__ void cvt_bf16_kernel(const float* __restrict__ src, bf16_t* __restrict__ dst,
                                size_t n) {
    size_t i = (size_t)blockIdx.x * blockDim.x + threadIdx.x;
    size_t stride = (size_t)gridDim.x * blockDim.x;
    for (; i < n; i += stride) dst[i] = (bf16_t)src[i];
}

// Gather strided query rows (bf16 copy): xqb[b, so, :] = xb[b, stride-sampled s, :]
__global__ void gather_xq_kernel(const bf16_t* __restrict__ xb, bf16_t* __restrict__ xqb) {
    size_t i = (size_t)blockIdx.x * blockDim.x + threadIdx.x;
    size_t total = (size_t)M_Q * C_IN;
    if (i >= total) return;
    int col = (int)(i % C_IN);
    int row = (int)(i / C_IN);
    int b  = row / SEQ_OUT;
    int so = row % SEQ_OUT;
    int s  = ((so / RES_OUT) * 2) * RES_IN + (so % RES_OUT) * 2;
    xqb[i] = xb[((size_t)b * SEQ + s) * C_IN + col];
}

// biasTab[h][q][k] = attn_biases[h, bias_idxs[q,k]]
__global__ void build_bias_kernel(const float* __restrict__ ab, const int* __restrict__ idxs,
                                  float* __restrict__ tab, int n_off) {
    size_t i = (size_t)blockIdx.x * blockDim.x + threadIdx.x;
    size_t total = (size_t)NUM_HEADS * SEQ_OUT * SEQ;
    if (i >= total) return;
    int h   = (int)(i / ((size_t)SEQ_OUT * SEQ));
    int rem = (int)(i % ((size_t)SEQ_OUT * SEQ));
    tab[i] = ab[h * n_off + idxs[rem]];
}

// Zero the key-padding columns 784..799 of vt[b][h][d][800]
__global__ void zero_vt_pad_kernel(bf16_t* __restrict__ vt) {
    size_t i = (size_t)blockIdx.x * blockDim.x + threadIdx.x;
    size_t total = (size_t)B_DIM * NUM_HEADS * D_V * (SEQ_P - SEQ);
    if (i >= total) return;
    size_t row = i / (SEQ_P - SEQ);
    int    j   = (int)(i % (SEQ_P - SEQ));
    vt[row * SEQ_P + SEQ + j] = (bf16_t)0.0f;
}

// ---------------------------------------------------------------------------
// Block GEMM core: C(128x128) += A(128xK) * B(128xK)^T, bf16 in, f32 acc.
// 256 threads = 8 waves arranged 4(M) x 2(N); each wave owns 2x4 16x16 tiles.
// A and B are both K-major (row-major [rows][K]).
// ---------------------------------------------------------------------------

#define LDST 40   // LDS row stride (bf16 elems), 80B = 5*16B aligned

static __device__ inline void gemm_core(const bf16_t* __restrict__ A,
                                        const bf16_t* __restrict__ B,
                                        int K, int m0, int n0,
                                        bf16_t* As, bf16_t* Bs,
                                        v8f c[2][4]) {
    const int tid   = threadIdx.x;
    const int lane  = tid & 31;
    const int wid   = tid >> 5;
    const int waveM = wid & 3;
    const int waveN = wid >> 2;
    const int lm    = lane & 15;
    const int lh    = lane >> 4;

    const int srow = tid >> 1;            // 0..127
    const int scol = (tid & 1) * 16;      // 0 or 16

    for (int k0 = 0; k0 < K; k0 += 32) {
        // Stage 128x32 A and B blocks into LDS (32B contiguous per thread each)
        {
            const u32x4* sa = (const u32x4*)(A + (size_t)(m0 + srow) * K + k0 + scol);
            u32x4* da = (u32x4*)(As + srow * LDST + scol);
            da[0] = sa[0]; da[1] = sa[1];
            const u32x4* sb = (const u32x4*)(B + (size_t)(n0 + srow) * K + k0 + scol);
            u32x4* db = (u32x4*)(Bs + srow * LDST + scol);
            db[0] = sb[0]; db[1] = sb[1];
        }
        __syncthreads();

        Frag a[2], bf[4];
        #pragma unroll
        for (int mt = 0; mt < 2; ++mt) {
            const bf16_t* p = As + (waveM * 32 + mt * 16 + lm) * LDST;
            a[mt].u[0] = *(const u32x4*)(p + lh * 8);
            a[mt].u[1] = *(const u32x4*)(p + 16 + lh * 8);
        }
        #pragma unroll
        for (int nt = 0; nt < 4; ++nt) {
            const bf16_t* p = Bs + (waveN * 64 + nt * 16 + lm) * LDST;
            bf[nt].u[0] = *(const u32x4*)(p + lh * 8);
            bf[nt].u[1] = *(const u32x4*)(p + 16 + lh * 8);
        }
        #pragma unroll
        for (int mt = 0; mt < 2; ++mt)
            #pragma unroll
            for (int nt = 0; nt < 4; ++nt)
                c[mt][nt] = wmma_bf16(a[mt], bf[nt], c[mt][nt]);
        __syncthreads();
    }
}

// ---------------------------------------------------------------------------
// GEMM 1: kv = BN(xb @ kv_w^T); scatter k -> kk[b][h][s][32] (bf16),
//         v -> vt[b][h][d][800] (bf16, transposed for P@V B-fragments).
// ---------------------------------------------------------------------------
__global__ __launch_bounds__(256)
void gemm_kv_kernel(const bf16_t* __restrict__ xb, const bf16_t* __restrict__ wb,
                    const float* __restrict__ scale, const float* __restrict__ bias,
                    bf16_t* __restrict__ kk, bf16_t* __restrict__ vt) {
    __shared__ bf16_t As[128 * LDST];
    __shared__ bf16_t Bs[128 * LDST];
    v8f c[2][4];
    #pragma unroll
    for (int mt = 0; mt < 2; ++mt)
        #pragma unroll
        for (int nt = 0; nt < 4; ++nt) c[mt][nt] = {};

    const int m0 = blockIdx.y * 128, n0 = blockIdx.x * 128;
    gemm_core(xb, wb, C_IN, m0, n0, As, Bs, c);

    const int lane = threadIdx.x & 31, wid = threadIdx.x >> 5;
    const int waveM = wid & 3, waveN = wid >> 2;
    const int lm = lane & 15, lh = lane >> 4;
    #pragma unroll
    for (int mt = 0; mt < 2; ++mt)
        #pragma unroll
        for (int nt = 0; nt < 4; ++nt)
            #pragma unroll
            for (int r = 0; r < 8; ++r) {
                int gm = m0 + waveM * 32 + mt * 16 + r + 8 * lh;
                int gn = n0 + waveN * 64 + nt * 16 + lm;
                float val = c[mt][nt][r] * scale[gn] + bias[gn];
                int b = gm / SEQ, s = gm % SEQ;
                int h = gn / (KEY_DIM + D_V), cc = gn % (KEY_DIM + D_V);
                if (cc < KEY_DIM) {
                    kk[(((size_t)(b * NUM_HEADS + h) * SEQ + s) * KEY_DIM) + cc] = (bf16_t)val;
                } else {
                    vt[(((size_t)(b * NUM_HEADS + h) * D_V + (cc - KEY_DIM)) * SEQ_P) + s] = (bf16_t)val;
                }
            }
}

// ---------------------------------------------------------------------------
// GEMM 2: q = BN(xqb @ q_w^T) * SCALE_ATT -> qb[b][h][so][32] (bf16)
// ---------------------------------------------------------------------------
__global__ __launch_bounds__(256)
void gemm_q_kernel(const bf16_t* __restrict__ xqb, const bf16_t* __restrict__ wb,
                   const float* __restrict__ scale, const float* __restrict__ bias,
                   bf16_t* __restrict__ qb) {
    __shared__ bf16_t As[128 * LDST];
    __shared__ bf16_t Bs[128 * LDST];
    v8f c[2][4];
    #pragma unroll
    for (int mt = 0; mt < 2; ++mt)
        #pragma unroll
        for (int nt = 0; nt < 4; ++nt) c[mt][nt] = {};

    const int m0 = blockIdx.y * 128, n0 = blockIdx.x * 128;
    gemm_core(xqb, wb, C_IN, m0, n0, As, Bs, c);

    const int lane = threadIdx.x & 31, wid = threadIdx.x >> 5;
    const int waveM = wid & 3, waveN = wid >> 2;
    const int lm = lane & 15, lh = lane >> 4;
    #pragma unroll
    for (int mt = 0; mt < 2; ++mt)
        #pragma unroll
        for (int nt = 0; nt < 4; ++nt)
            #pragma unroll
            for (int r = 0; r < 8; ++r) {
                int gm = m0 + waveM * 32 + mt * 16 + r + 8 * lh;
                int gn = n0 + waveN * 64 + nt * 16 + lm;
                float val = (c[mt][nt][r] * scale[gn] + bias[gn]) * SCALE_ATT;
                int b = gm / SEQ_OUT, so = gm % SEQ_OUT;
                int h = gn / KEY_DIM, d = gn % KEY_DIM;
                qb[(((size_t)(b * NUM_HEADS + h) * SEQ_OUT + so) * KEY_DIM) + d] = (bf16_t)val;
            }
}

// ---------------------------------------------------------------------------
// Fused attention: per (qt, h, b), one wave.
//   S = q k^T + bias (bias via WMMA C operand), softmax over 784 keys,
//   O = hardswish(P V) -> Ob[b*196+q][h*128+d] (bf16)
// ---------------------------------------------------------------------------
__global__ __launch_bounds__(32)
void attn_kernel(const bf16_t* __restrict__ qb, const bf16_t* __restrict__ kk,
                 const bf16_t* __restrict__ vt, const float* __restrict__ biasTab,
                 bf16_t* __restrict__ Ob) {
    __shared__ float S[16][SEQ_P];   // 51200 B: scores, then probabilities in place

    const int qt = blockIdx.x;
    const int h  = blockIdx.y;
    const int b  = blockIdx.z;
    const int lane = threadIdx.x;
    const int lm = lane & 15, lh = lane >> 4;

    const bf16_t* qbase = qb + ((size_t)(b * NUM_HEADS + h) * SEQ_OUT) * KEY_DIM;
    const bf16_t* kbase = kk + ((size_t)(b * NUM_HEADS + h) * SEQ) * KEY_DIM;
    const bf16_t* vbase = vt + ((size_t)(b * NUM_HEADS + h) * D_V) * SEQ_P;

    // ---- Phase 1: S = q k^T + bias (49 WMMAs, K-contraction = 32 = one WMMA)
    const int arow = min(qt * 16 + lm, SEQ_OUT - 1);
    Frag a;
    {
        const bf16_t* ap = qbase + (size_t)arow * KEY_DIM;
        a.u[0] = *(const u32x4*)(ap + lh * 8);
        a.u[1] = *(const u32x4*)(ap + 16 + lh * 8);
    }
    for (int kt = 0; kt < SEQ / 16; ++kt) {
        Frag bf;
        const bf16_t* kp = kbase + (size_t)(kt * 16 + lm) * KEY_DIM;
        bf.u[0] = *(const u32x4*)(kp + lh * 8);
        bf.u[1] = *(const u32x4*)(kp + 16 + lh * 8);

        v8f c;
        #pragma unroll
        for (int r = 0; r < 8; ++r) {
            int qrow = min(qt * 16 + r + 8 * lh, SEQ_OUT - 1);
            c[r] = biasTab[((size_t)h * SEQ_OUT + qrow) * SEQ + kt * 16 + lm];
        }
        c = wmma_bf16(a, bf, c);
        #pragma unroll
        for (int r = 0; r < 8; ++r)
            S[r + 8 * lh][kt * 16 + lm] = c[r];
    }
    __syncthreads();

    // ---- Phase 2: softmax rows; each lane handles half a row (392 cols)
    {
        const int n0h = lh * (SEQ / 2);
        float mx = -3.0e38f;
        for (int i = 0; i < SEQ / 2; ++i) mx = fmaxf(mx, S[lm][n0h + i]);
        mx = fmaxf(mx, __shfl_xor(mx, 16, 32));
        float sum = 0.0f;
        for (int i = 0; i < SEQ / 2; ++i) {
            float e = __expf(S[lm][n0h + i] - mx);
            sum += e;
            S[lm][n0h + i] = e;
        }
        sum += __shfl_xor(sum, 16, 32);
        float inv = 1.0f / sum;
        for (int i = 0; i < SEQ / 2; ++i) S[lm][n0h + i] *= inv;
        // zero the key padding 784..799
        #pragma unroll
        for (int j = 0; j < 8; ++j) S[lm][SEQ + lh * 8 + j] = 0.0f;
    }
    __syncthreads();

    // ---- Phase 3: O = P @ V  (25 k-steps x 8 d-tiles = 200 WMMAs)
    v8f acc[8];
    #pragma unroll
    for (int dt = 0; dt < 8; ++dt) acc[dt] = {};

    for (int kt2 = 0; kt2 < SEQ_P / 32; ++kt2) {
        const int k0 = kt2 * 32;
        Frag pa;
        #pragma unroll
        for (int j = 0; j < 8; ++j) {
            pa.e[j]     = (bf16_t)S[lm][k0 + lh * 8 + j];
            pa.e[8 + j] = (bf16_t)S[lm][k0 + 16 + lh * 8 + j];
        }
        #pragma unroll
        for (int dt = 0; dt < 8; ++dt) {
            Frag vb;
            const bf16_t* vp = vbase + (size_t)(dt * 16 + lm) * SEQ_P + k0;
            vb.u[0] = *(const u32x4*)(vp + lh * 8);
            vb.u[1] = *(const u32x4*)(vp + 16 + lh * 8);
            acc[dt] = wmma_bf16(pa, vb, acc[dt]);
        }
    }

    // ---- Epilogue: hardswish, masked bf16 store to Ob[b*196+q][h*128+d]
    #pragma unroll
    for (int dt = 0; dt < 8; ++dt)
        #pragma unroll
        for (int r = 0; r < 8; ++r) {
            int qrow = qt * 16 + r + 8 * lh;
            if (qrow < SEQ_OUT) {
                float x = acc[dt][r];
                float hs = x * fminf(fmaxf(x + 3.0f, 0.0f), 6.0f) * (1.0f / 6.0f);
                Ob[((size_t)(b * SEQ_OUT + qrow)) * OUT_PROJ + h * D_V + dt * 16 + lm] = (bf16_t)hs;
            }
        }
}

// ---------------------------------------------------------------------------
// GEMM 3: out = BN(Ob @ proj_w^T) -> f32 [12544][768]
// ---------------------------------------------------------------------------
__global__ __launch_bounds__(256)
void gemm_proj_kernel(const bf16_t* __restrict__ Ob, const bf16_t* __restrict__ wb,
                      const float* __restrict__ scale, const float* __restrict__ bias,
                      float* __restrict__ out) {
    __shared__ bf16_t As[128 * LDST];
    __shared__ bf16_t Bs[128 * LDST];
    v8f c[2][4];
    #pragma unroll
    for (int mt = 0; mt < 2; ++mt)
        #pragma unroll
        for (int nt = 0; nt < 4; ++nt) c[mt][nt] = {};

    const int m0 = blockIdx.y * 128, n0 = blockIdx.x * 128;
    gemm_core(Ob, wb, OUT_PROJ, m0, n0, As, Bs, c);

    const int lane = threadIdx.x & 31, wid = threadIdx.x >> 5;
    const int waveM = wid & 3, waveN = wid >> 2;
    const int lm = lane & 15, lh = lane >> 4;
    #pragma unroll
    for (int mt = 0; mt < 2; ++mt)
        #pragma unroll
        for (int nt = 0; nt < 4; ++nt)
            #pragma unroll
            for (int r = 0; r < 8; ++r) {
                int gm = m0 + waveM * 32 + mt * 16 + r + 8 * lh;
                int gn = n0 + waveN * 64 + nt * 16 + lm;
                float val = c[mt][nt][r] * scale[gn] + bias[gn];
                out[(size_t)gm * C_OUT + gn] = val;
            }
}

// ---------------------------------------------------------------------------
// Host launch
// ---------------------------------------------------------------------------
extern "C" void kernel_launch(void* const* d_in, const int* in_sizes, int n_in,
                              void* d_out, int out_size, void* d_ws, size_t ws_size,
                              hipStream_t stream) {
    const float* x    = (const float*)d_in[0];
    const float* kv_w = (const float*)d_in[1];
    const float* kv_g = (const float*)d_in[2];
    const float* kv_b = (const float*)d_in[3];
    const float* kv_m = (const float*)d_in[4];
    const float* kv_v = (const float*)d_in[5];
    const float* q_w  = (const float*)d_in[6];
    const float* q_g  = (const float*)d_in[7];
    const float* q_b  = (const float*)d_in[8];
    const float* q_m  = (const float*)d_in[9];
    const float* q_v  = (const float*)d_in[10];
    const float* p_w  = (const float*)d_in[11];
    const float* p_g  = (const float*)d_in[12];
    const float* p_b  = (const float*)d_in[13];
    const float* p_m  = (const float*)d_in[14];
    const float* p_v  = (const float*)d_in[15];
    const float* attn_biases = (const float*)d_in[16];
    const int*   bias_idxs   = (const int*)d_in[17];
    const int n_off = in_sizes[16] / NUM_HEADS;
    float* out = (float*)d_out;

    // workspace carve-up (256B aligned bump allocator)
    char*  base = (char*)d_ws;
    size_t off  = 0;
    auto alloc = [&](size_t bytes) -> void* {
        void* p = base + off;
        off += (bytes + 255) & ~(size_t)255;
        return p;
    };
    bf16_t* xb   = (bf16_t*)alloc((size_t)M_KV * C_IN * 2);
    bf16_t* xqb  = (bf16_t*)alloc((size_t)M_Q * C_IN * 2);
    bf16_t* kvwb = (bf16_t*)alloc((size_t)OUT_KV * C_IN * 2);
    bf16_t* qwb  = (bf16_t*)alloc((size_t)OUT_Q * C_IN * 2);
    bf16_t* pwb  = (bf16_t*)alloc((size_t)C_OUT * OUT_PROJ * 2);
    float*  kvS  = (float*)alloc(OUT_KV * 4);
    float*  kvB  = (float*)alloc(OUT_KV * 4);
    float*  qS   = (float*)alloc(OUT_Q * 4);
    float*  qB   = (float*)alloc(OUT_Q * 4);
    float*  pS   = (float*)alloc(C_OUT * 4);
    float*  pB   = (float*)alloc(C_OUT * 4);
    bf16_t* kk   = (bf16_t*)alloc((size_t)B_DIM * NUM_HEADS * SEQ * KEY_DIM * 2);
    bf16_t* vt   = (bf16_t*)alloc((size_t)B_DIM * NUM_HEADS * D_V * SEQ_P * 2);
    bf16_t* qb   = (bf16_t*)alloc((size_t)B_DIM * NUM_HEADS * SEQ_OUT * KEY_DIM * 2);
    float*  biasTab = (float*)alloc((size_t)NUM_HEADS * SEQ_OUT * SEQ * 4);
    bf16_t* Ob   = (bf16_t*)alloc((size_t)M_Q * OUT_PROJ * 2);
    (void)ws_size; (void)n_in; (void)out_size;

    // --- prep
    bn_fuse_kernel<<<(OUT_KV + 255) / 256, 256, 0, stream>>>(kv_g, kv_b, kv_m, kv_v, kvS, kvB, OUT_KV);
    bn_fuse_kernel<<<(OUT_Q + 255) / 256, 256, 0, stream>>>(q_g, q_b, q_m, q_v, qS, qB, OUT_Q);
    bn_fuse_kernel<<<(C_OUT + 255) / 256, 256, 0, stream>>>(p_g, p_b, p_m, p_v, pS, pB, C_OUT);

    {
        size_t n;
        n = (size_t)OUT_KV * C_IN;
        cvt_bf16_kernel<<<(unsigned)((n + 255) / 256), 256, 0, stream>>>(kv_w, kvwb, n);
        n = (size_t)OUT_Q * C_IN;
        cvt_bf16_kernel<<<(unsigned)((n + 255) / 256), 256, 0, stream>>>(q_w, qwb, n);
        n = (size_t)C_OUT * OUT_PROJ;
        cvt_bf16_kernel<<<(unsigned)((n + 255) / 256), 256, 0, stream>>>(p_w, pwb, n);
        n = (size_t)M_KV * C_IN;
        cvt_bf16_kernel<<<(unsigned)((n + 255) / 256), 256, 0, stream>>>(x, xb, n);
    }
    {
        size_t n = (size_t)M_Q * C_IN;
        gather_xq_kernel<<<(unsigned)((n + 255) / 256), 256, 0, stream>>>(xb, xqb);
    }
    {
        size_t n = (size_t)NUM_HEADS * SEQ_OUT * SEQ;
        build_bias_kernel<<<(unsigned)((n + 255) / 256), 256, 0, stream>>>(attn_biases, bias_idxs, biasTab, n_off);
    }
    {
        size_t n = (size_t)B_DIM * NUM_HEADS * D_V * (SEQ_P - SEQ);
        zero_vt_pad_kernel<<<(unsigned)((n + 255) / 256), 256, 0, stream>>>(vt);
    }

    // --- GEMM kv: M=50176, N=2560, K=512
    gemm_kv_kernel<<<dim3(OUT_KV / 128, M_KV / 128), 256, 0, stream>>>(xb, kvwb, kvS, kvB, kk, vt);
    // --- GEMM q: M=12544, N=512, K=512
    gemm_q_kernel<<<dim3(OUT_Q / 128, M_Q / 128), 256, 0, stream>>>(xqb, qwb, qS, qB, qb);
    // --- fused attention
    attn_kernel<<<dim3(QT_TILES, NUM_HEADS, B_DIM), 32, 0, stream>>>(qb, kk, vt, biasTab, Ob);
    // --- GEMM proj: M=12544, N=768, K=2048
    gemm_proj_kernel<<<dim3(C_OUT / 128, M_Q / 128), 256, 0, stream>>>(Ob, pwb, pS, pB, out);
}
